// ResidualSovnet_11072425689233
// MI455X (gfx1250) — compile-verified, matrix operand order checked by reading, not run on hardware
//
#include <hip/hip_runtime.h>
#include <hip/hip_bf16.h>
#include <stdint.h>

// ---------------------------------------------------------------------------
// CDNA5 WMMA types
// ---------------------------------------------------------------------------
typedef __attribute__((ext_vector_type(16))) __bf16 v16bf;
typedef __attribute__((ext_vector_type(8)))  float  v8f;
typedef __attribute__((ext_vector_type(4)))  unsigned int u32x4;

union Frag { v16bf v; u32x4 q[2]; };

__device__ __forceinline__ unsigned short f32_to_bf16(float f) {
  union { float f; unsigned u; } x;
  x.f = f;
  unsigned r = x.u + 0x7FFFu + ((x.u >> 16) & 1u);   // round-to-nearest-even
  return (unsigned short)(r >> 16);
}

// ---------------------------------------------------------------------------
// Weight transform: w (Cout, Cin, S, ks, ks) f32 ->
//   Amat bf16 row-major [M=4*Cout][Kpad], m = r*Cout+co,
//   k = (c*S+s)*ks*ks + ky*ks + kx
//   value = w[co][c][(s-r)%4][rot90^r(ky,kx)]   (S==4 p4conv; S==1 z2 conv)
// ---------------------------------------------------------------------------
__global__ void k_weight(const float* __restrict__ w, unsigned short* __restrict__ Amat,
                         int Cout, int Cin, int S, int ks, int Kpad, int Keff) {
  long M = 4L * Cout;
  long i = (long)blockIdx.x * blockDim.x + threadIdx.x;
  if (i >= M * Kpad) return;
  int k = (int)(i % Kpad);
  int m = (int)(i / Kpad);
  int r = m / Cout;
  int co = m % Cout;
  float v = 0.f;
  if (k < Keff) {
    int kx = k % ks;
    int t = k / ks;
    int ky = t % ks;
    int ce = t / ks;
    int c = ce / S;
    int sIdx = ce % S;
    int sSrc = (S == 4) ? ((sIdx - r) & 3) : sIdx;
    int e = ks - 1;
    int sy, sx;
    switch (r & 3) {
      case 0:  sy = ky;      sx = kx;      break;
      case 1:  sy = kx;      sx = e - ky;  break;
      case 2:  sy = e - ky;  sx = e - kx;  break;
      default: sy = e - kx;  sx = ky;      break;
    }
    v = w[(((long)co * Cin + c) * S + sSrc) * ks * ks + sy * ks + sx];
  }
  Amat[(long)m * Kpad + k] = f32_to_bf16(v);
}

// ---------------------------------------------------------------------------
// im2col: input (Bb, CinEff, H, W) f32 -> Bp bf16 in K-PANEL layout:
//   Bp[ ((k/32)*N + n)*32 + (k%32) ],  N = Bb*Hp*Wp
// This makes each lane's WMMA B-fragment two contiguous 16B chunks.
// ---------------------------------------------------------------------------
__global__ void k_im2col(const float* __restrict__ in, unsigned short* __restrict__ Bp,
                         int CinEff, int H, int W, int Hp, int Wp, int Bb,
                         int stride, int pad, int ks, int Kpad, int Keff) {
  long N = (long)Bb * Hp * Wp;
  long total = (long)Kpad * N;
  long i = (long)blockIdx.x * blockDim.x + threadIdx.x;
  if (i >= total) return;
  int k = (int)(i / N);
  long n = i - (long)k * N;
  float v = 0.f;
  if (k < Keff) {
    int kx = k % ks;
    int t = k / ks;
    int ky = t % ks;
    int ce = t / ks;
    int xo = (int)(n % Wp);
    long t2 = n / Wp;
    int yo = (int)(t2 % Hp);
    int b = (int)(t2 / Hp);
    int yi = yo * stride - pad + ky;
    int xi = xo * stride - pad + kx;
    if (yi >= 0 && yi < H && xi >= 0 && xi < W)
      v = in[((long)b * CinEff + ce) * H * W + (long)yi * W + xi];
  }
  Bp[((long)(k >> 5) * N + n) * 32 + (k & 31)] = f32_to_bf16(v);
}

// ---------------------------------------------------------------------------
// WMMA GEMM: C[M][N] f32 = A[M][K](bf16, row-major) * B[K][N](bf16, K-panel)
// All M, N in this net are multiples of 64; K is a multiple of 32 with
// zero-padded tails -> NO bounds checks anywhere.
// Block = 128 threads = 4 waves (2x2); wave tile = 32x32 = 2x2 WMMA tiles.
// Fragments are loaded straight from global with b128 loads (ISA 16-bit A/B
// layout is K-contiguous per lane: kk = {half*8..+7} and {16+half*8..+7}).
// ---------------------------------------------------------------------------
__global__ __launch_bounds__(128)
void k_gemm_wmma(const unsigned short* __restrict__ A,
                 const unsigned short* __restrict__ Bp,
                 float* __restrict__ C, int N, int K) {
  const int lane = threadIdx.x & 31;
  const int wave = threadIdx.x >> 5;
  const int wm = wave >> 1, wn = wave & 1;
  const int bm = blockIdx.y * 64 + wm * 32;
  const int bn = blockIdx.x * 64 + wn * 32;
  const int half = lane >> 4;
  const int l16  = lane & 15;

  v8f acc00 = {}, acc01 = {}, acc10 = {}, acc11 = {};

  const unsigned short* a0p = A + (long)(bm + l16) * K + half * 8;
  const unsigned short* a1p = a0p + 16L * K;
  const unsigned short* b0p = Bp + (long)(bn + l16) * 32 + half * 8;
  const unsigned short* b1p = b0p + 16L * 32;
  const long panelStride = (long)N * 32;

  for (int k0 = 0; k0 < K; k0 += 32) {
    Frag a0, a1, b0, b1;
    a0.q[0] = *(const u32x4*)(a0p);
    a0.q[1] = *(const u32x4*)(a0p + 16);
    a1.q[0] = *(const u32x4*)(a1p);
    a1.q[1] = *(const u32x4*)(a1p + 16);
    b0.q[0] = *(const u32x4*)(b0p);
    b0.q[1] = *(const u32x4*)(b0p + 16);
    b1.q[0] = *(const u32x4*)(b1p);
    b1.q[1] = *(const u32x4*)(b1p + 16);
    if (k0 + 32 < K) {                      // speculative prefetch of next slab
      __builtin_prefetch(a0p + 32, 0, 1);
      __builtin_prefetch(b0p + panelStride, 0, 1);
    }
    acc00 = __builtin_amdgcn_wmma_f32_16x16x32_bf16(false, a0.v, false, b0.v,
                                                    (short)0, acc00, false, false);
    acc01 = __builtin_amdgcn_wmma_f32_16x16x32_bf16(false, a0.v, false, b1.v,
                                                    (short)0, acc01, false, false);
    acc10 = __builtin_amdgcn_wmma_f32_16x16x32_bf16(false, a1.v, false, b0.v,
                                                    (short)0, acc10, false, false);
    acc11 = __builtin_amdgcn_wmma_f32_16x16x32_bf16(false, a1.v, false, b1.v,
                                                    (short)0, acc11, false, false);
    a0p += 32; a1p += 32;
    b0p += panelStride; b1p += panelStride;
  }

  // C/D layout: VGPR v -> M = tile + 8*half + v, N = tile + l16
  const int rbase = bm + 8 * half;
  const int cbase = bn + l16;
#pragma unroll
  for (int v = 0; v < 8; ++v) {
    C[(long)(rbase + v) * N + cbase]            = acc00[v];
    C[(long)(rbase + v) * N + cbase + 16]       = acc01[v];
    C[(long)(rbase + 16 + v) * N + cbase]       = acc10[v];
    C[(long)(rbase + 16 + v) * N + cbase + 16]  = acc11[v];
  }
}

// ---------------------------------------------------------------------------
// Unpack GEMM output C[M=4*Cout][N=Bb*HW] -> act (Bb, Cout, 4, HW) (+bias)
// ---------------------------------------------------------------------------
__global__ void k_unpack(const float* __restrict__ Cg, const float* __restrict__ bias,
                         float* __restrict__ act, int Cout, int HW, int Bb, int N) {
  long total = (long)Bb * Cout * 4 * HW;
  long i = (long)blockIdx.x * blockDim.x + threadIdx.x;
  if (i >= total) return;
  int yx = (int)(i % HW);
  long t = i / HW;
  int r = (int)(t % 4); t /= 4;
  int co = (int)(t % Cout);
  int b  = (int)(t / Cout);
  int m = r * Cout + co;
  long n = (long)b * HW + yx;
  float v = Cg[(long)m * N + n];
  if (bias) v += bias[co];
  act[i] = v;
}

// ---------------------------------------------------------------------------
// BatchNorm3d (train mode, gamma=1 beta=0): stats over (B,4,H,W) per channel
// ---------------------------------------------------------------------------
__global__ void k_bn_stats(const float* __restrict__ x, float* __restrict__ mean,
                           float* __restrict__ invstd, int Bb, int Cc, int RHW,
                           float eps) {
  int c = blockIdx.x;
  int tid = threadIdx.x;
  long total = (long)Bb * RHW;
  float s = 0.f, s2 = 0.f;
  for (long j = tid; j < total; j += blockDim.x) {
    int b = (int)(j / RHW);
    int rm = (int)(j % RHW);
    float v = x[((long)b * Cc + c) * RHW + rm];
    s += v; s2 += v * v;
  }
  __shared__ float sh[256], sh2[256];
  sh[tid] = s; sh2[tid] = s2;
  __syncthreads();
  for (int off = 128; off > 0; off >>= 1) {
    if (tid < off) { sh[tid] += sh[tid + off]; sh2[tid] += sh2[tid + off]; }
    __syncthreads();
  }
  if (tid == 0) {
    float inv = 1.0f / (float)total;
    float m = sh[0] * inv;
    float v = sh2[0] * inv - m * m;
    mean[c] = m;
    invstd[c] = rsqrtf(v + eps);
  }
}

__global__ void k_bn_apply(const float* __restrict__ x, float* __restrict__ y,
                           const float* __restrict__ mean, const float* __restrict__ invstd,
                           const float* __restrict__ res, int Cc, int RHW, long total,
                           int relu) {
  long i = (long)blockIdx.x * blockDim.x + threadIdx.x;
  if (i >= total) return;
  int c = (int)((i / RHW) % Cc);
  float v = (x[i] - mean[c]) * invstd[c];
  if (res) v += res[i];
  if (relu) v = fmaxf(v, 0.f);
  y[i] = v;
}

// ---------------------------------------------------------------------------
// Per-batch LayerNorm (primary caps): stats over all (C,4,H,W) per batch b
// ---------------------------------------------------------------------------
__global__ void k_ln_stats(const float* __restrict__ x, float* __restrict__ mean,
                           float* __restrict__ invstd, long per, float eps) {
  int b = blockIdx.x;
  int tid = threadIdx.x;
  const float* xp = x + (long)b * per;
  float s = 0.f, s2 = 0.f;
  for (long j = tid; j < per; j += blockDim.x) {
    float v = xp[j];
    s += v; s2 += v * v;
  }
  __shared__ float sh[256], sh2[256];
  sh[tid] = s; sh2[tid] = s2;
  __syncthreads();
  for (int off = 128; off > 0; off >>= 1) {
    if (tid < off) { sh[tid] += sh[tid + off]; sh2[tid] += sh2[tid + off]; }
    __syncthreads();
  }
  if (tid == 0) {
    float inv = 1.0f / (float)per;
    float m = sh[0] * inv;
    float v = sh2[0] * inv - m * m;
    mean[b] = m;
    invstd[b] = rsqrtf(v + eps);
  }
}

__global__ void k_ln_apply(const float* __restrict__ x, float* __restrict__ y,
                           const float* __restrict__ mean, const float* __restrict__ invstd,
                           long per, long total) {
  long i = (long)blockIdx.x * blockDim.x + threadIdx.x;
  if (i >= total) return;
  int b = (int)(i / per);
  y[i] = (x[i] - mean[b]) * invstd[b];
}

// ---------------------------------------------------------------------------
// Capsule routing. One wave32 per (b, o, r, yx); lane i = input capsule.
// Reads the conv GEMM output directly:
//   p_raw[i][d] = Cg[(r*CO + o*16 + d)*Ncols + (b*32+i)*HW + yx] + bias[o*16+d]
// Row LayerNorm over d; row-normalized similarity sum collapses to
// (p_i . q)/||p_i||^2 with q = sum_j p_j; softmax over i; s = sum_i score_i p_i.
// Output layout (B, n_out, 16, 4, HW).
// ---------------------------------------------------------------------------
__global__ __launch_bounds__(32)
void k_routing(const float* __restrict__ Cg, const float* __restrict__ bias,
               float* __restrict__ out, int n_out, int B, int HW) {
  long loc = blockIdx.x;
  int yx = (int)(loc % HW); loc /= HW;
  int r = (int)(loc % 4);   loc /= 4;
  int o = (int)(loc % n_out);
  int b = (int)(loc / n_out);
  int i = threadIdx.x;   // input capsule index, n_in == 32

  __shared__ float sp[32][17];
  __shared__ float qv[16];
  __shared__ float ssc[32];

  long Ncols = (long)B * 32 * HW;
  long n = (long)(b * 32 + i) * HW + yx;
  int CO = n_out * 16;

  float p[16];
  float mu = 0.f;
#pragma unroll
  for (int d = 0; d < 16; ++d) {
    float v = Cg[((long)r * CO + o * 16 + d) * Ncols + n] + bias[o * 16 + d];
    p[d] = v; mu += v;
  }
  mu *= (1.0f / 16.0f);
  float var = 0.f;
#pragma unroll
  for (int d = 0; d < 16; ++d) { float t = p[d] - mu; var += t * t; }
  var *= (1.0f / 16.0f);
  float is = rsqrtf(var + 1e-5f);
  float nsq = 0.f;
#pragma unroll
  for (int d = 0; d < 16; ++d) {
    p[d] = (p[d] - mu) * is;
    nsq += p[d] * p[d];
    sp[i][d] = p[d];
  }
  __syncthreads();
  if (i < 16) {
    float q = 0.f;
    for (int j = 0; j < 32; ++j) q += sp[j][i];
    qv[i] = q;
  }
  __syncthreads();
  float dot = 0.f;
#pragma unroll
  for (int d = 0; d < 16; ++d) dot += p[d] * qv[d];
  float rowsum = dot / fmaxf(nsq, 1e-8f);
  // softmax over the 32 lanes (wave32 shuffles)
  float mx = rowsum;
  for (int off = 16; off > 0; off >>= 1) mx = fmaxf(mx, __shfl_xor(mx, off, 32));
  float e = __expf(rowsum - mx);
  float se = e;
  for (int off = 16; off > 0; off >>= 1) se += __shfl_xor(se, off, 32);
  ssc[i] = e / se;
  __syncthreads();
  if (i < 16) {
    float s = 0.f;
    for (int j = 0; j < 32; ++j) s += ssc[j] * sp[j][i];
    long oidx = (((long)(b * n_out + o) * 16 + i) * 4 + r) * HW + yx;
    out[oidx] = s;
  }
}

// squash over rotation axis: layout (B, n_out, 16, 4, HW), in-place
__global__ void k_squash(float* __restrict__ s, long total, int HW) {
  long i = (long)blockIdx.x * blockDim.x + threadIdx.x;   // over (b,o,d,yx)
  if (i >= total) return;
  int yx = (int)(i % HW);
  long cd = i / HW;
  long base = cd * 4 * HW + yx;
  float v0 = s[base], v1 = s[base + HW], v2 = s[base + 2L * HW], v3 = s[base + 3L * HW];
  float nsq2 = v0 * v0 + v1 * v1 + v2 * v2 + v3 * v3;
  float f = (nsq2 / (1.0f + nsq2)) * rsqrtf(nsq2 + 1e-16f);
  s[base] = v0 * f;
  s[base + HW] = v1 * f;
  s[base + 2L * HW] = v2 * f;
  s[base + 3L * HW] = v3 * f;
}

// final: caps (B,10,16,4) -> out[b*10+o] = max_r ||caps[b,o,:,r]||
__global__ void k_final(const float* __restrict__ caps, float* __restrict__ out, int B) {
  int i = blockIdx.x * blockDim.x + threadIdx.x;
  if (i >= B * 10) return;
  const float* c = caps + (long)i * 64;
  float best = -1e30f;
  for (int r = 0; r < 4; ++r) {
    float ss = 0.f;
    for (int d = 0; d < 16; ++d) { float v = c[d * 4 + r]; ss += v * v; }
    best = fmaxf(best, sqrtf(ss));
  }
  out[i] = best;
}

// ---------------------------------------------------------------------------
// Host-side orchestration
// ---------------------------------------------------------------------------
static inline int cdiv(long a, long b) { return (int)((a + b - 1) / b); }

static void run_conv(hipStream_t stream, const float* w, const float* bias,
                     const float* in, float* outAct,
                     int Bb, int Cin, int S, int Cout,
                     int H, int W, int stride, int pad, int ks,
                     unsigned short* Amat, unsigned short* Bmat, float* Cbuf,
                     int* HpOut, int* WpOut) {
  int Hp = (H + 2 * pad - ks) / stride + 1;
  int Wp = (W + 2 * pad - ks) / stride + 1;
  int CinEff = Cin * S;
  int Keff = CinEff * ks * ks;
  int Kpad = (Keff + 31) & ~31;
  int M = 4 * Cout;              // always a multiple of 64 in this network
  long N = (long)Bb * Hp * Wp;   // always a multiple of 64 in this network

  k_weight<<<cdiv((long)M * Kpad, 256), 256, 0, stream>>>(w, Amat, Cout, Cin, S, ks, Kpad, Keff);
  k_im2col<<<cdiv((long)Kpad * N, 256), 256, 0, stream>>>(in, Bmat, CinEff, H, W, Hp, Wp,
                                                          Bb, stride, pad, ks, Kpad, Keff);
  dim3 g((unsigned)(N / 64), (unsigned)(M / 64));
  k_gemm_wmma<<<g, 128, 0, stream>>>(Amat, Bmat, Cbuf, (int)N, Kpad);
  if (outAct) {
    long uTot = (long)Bb * Cout * 4 * Hp * Wp;
    k_unpack<<<cdiv(uTot, 256), 256, 0, stream>>>(Cbuf, bias, outAct, Cout, Hp * Wp, Bb, (int)N);
  }
  if (HpOut) *HpOut = Hp;
  if (WpOut) *WpOut = Wp;
}

static void run_bn(hipStream_t stream, const float* xin, float* y, float* mean, float* invstd,
                   int Bb, int Cc, int RHW, const float* res, int relu) {
  k_bn_stats<<<Cc, 256, 0, stream>>>(xin, mean, invstd, Bb, Cc, RHW, 1e-5f);
  long tot = (long)Bb * Cc * RHW;
  k_bn_apply<<<cdiv(tot, 256), 256, 0, stream>>>(xin, y, mean, invstd, res, Cc, RHW, tot, relu);
}

extern "C" void kernel_launch(void* const* d_in, const int* in_sizes, int n_in,
                              void* d_out, int out_size, void* d_ws, size_t ws_size,
                              hipStream_t stream) {
  (void)in_sizes; (void)n_in; (void)out_size; (void)ws_size;

  // ---- inputs (setup_inputs flattening order) ----
  int idx = 0;
  const float* x       = (const float*)d_in[idx++];   // (8,3,32,32)
  const float* w_conv0 = (const float*)d_in[idx++];   // (64,3,3,3)
  const float* res1w[3][2];
  for (int i = 0; i < 3; ++i) {
    res1w[i][0] = (const float*)d_in[idx++];
    res1w[i][1] = (const float*)d_in[idx++];
  }
  const float* r2_w1[4]; const float* r2_w2[4]; const float* r2_ws;
  r2_w1[0] = (const float*)d_in[idx++];
  r2_w2[0] = (const float*)d_in[idx++];
  r2_ws    = (const float*)d_in[idx++];
  for (int i = 1; i < 4; ++i) {
    r2_w1[i] = (const float*)d_in[idx++];
    r2_w2[i] = (const float*)d_in[idx++];
  }
  const float* w_primary = (const float*)d_in[idx++];
  const float* b_primary = (const float*)d_in[idx++];
  const float* caps_w[5]; const float* caps_b[5];
  for (int i = 0; i < 5; ++i) {
    caps_w[i] = (const float*)d_in[idx++];
    caps_b[i] = (const float*)d_in[idx++];
  }
  float* out = (float*)d_out;

  // ---- workspace carve ----
  const size_t MB = 1024 * 1024;
  char* base = (char*)d_ws;
  float*          actA  = (float*)(base + 0 * MB);     // 20 MB
  float*          actB  = (float*)(base + 20 * MB);    // 20 MB
  float*          actC  = (float*)(base + 40 * MB);    // 20 MB
  float*          actD  = (float*)(base + 60 * MB);    // 20 MB
  unsigned short* Amat  = (unsigned short*)(base + 80 * MB);   // 20 MB bf16
  unsigned short* Bmat  = (unsigned short*)(base + 100 * MB);  // 40 MB bf16
  float*          Cbuf  = (float*)(base + 140 * MB);   // 104 MB
  float*          stats  = (float*)(base + 246 * MB);  // means
  float*          stats2 = stats + 1024;               // invstds

  const int B = 8;
  int Hp, Wp;

  // ---- conv0 (z2) + BN + ReLU : (8,64,4,32,32) ----
  run_conv(stream, w_conv0, nullptr, x, actA, B, 3, 1, 64, 32, 32, 1, 1, 3,
           Amat, Bmat, Cbuf, &Hp, &Wp);
  run_bn(stream, actA, actB, stats, stats2, B, 64, 4 * 32 * 32, nullptr, 1);

  // ---- res1 x3 (identity shortcut, C=64, 32x32) ----
  for (int i = 0; i < 3; ++i) {
    run_conv(stream, res1w[i][0], nullptr, actB, actA, B, 64, 4, 64, 32, 32, 1, 1, 3,
             Amat, Bmat, Cbuf, nullptr, nullptr);
    run_bn(stream, actA, actA, stats, stats2, B, 64, 4 * 1024, nullptr, 1);
    run_conv(stream, res1w[i][1], nullptr, actA, actC, B, 64, 4, 64, 32, 32, 1, 1, 3,
             Amat, Bmat, Cbuf, nullptr, nullptr);
    run_bn(stream, actC, actB, stats, stats2, B, 64, 4 * 1024, actB, 1);  // relu(bn+sc)
  }

  // ---- res2[0] (stride-2, 1x1 shortcut, 64->128, 32->16) ----
  run_conv(stream, r2_w1[0], nullptr, actB, actA, B, 64, 4, 128, 32, 32, 2, 1, 3,
           Amat, Bmat, Cbuf, &Hp, &Wp);                              // 16x16
  run_bn(stream, actA, actA, stats, stats2, B, 128, 4 * 256, nullptr, 1);
  run_conv(stream, r2_w2[0], nullptr, actA, actC, B, 128, 4, 128, 16, 16, 1, 1, 3,
           Amat, Bmat, Cbuf, nullptr, nullptr);
  run_conv(stream, r2_ws, nullptr, actB, actD, B, 64, 4, 128, 32, 32, 2, 0, 1,
           Amat, Bmat, Cbuf, nullptr, nullptr);
  run_bn(stream, actD, actD, stats, stats2, B, 128, 4 * 256, nullptr, 0);  // bn(shortcut)
  run_bn(stream, actC, actB, stats, stats2, B, 128, 4 * 256, actD, 1);     // relu(bn+sc)

  // ---- res2[1..3] (identity, C=128, 16x16) ----
  for (int i = 1; i < 4; ++i) {
    run_conv(stream, r2_w1[i], nullptr, actB, actA, B, 128, 4, 128, 16, 16, 1, 1, 3,
             Amat, Bmat, Cbuf, nullptr, nullptr);
    run_bn(stream, actA, actA, stats, stats2, B, 128, 4 * 256, nullptr, 1);
    run_conv(stream, r2_w2[i], nullptr, actA, actC, B, 128, 4, 128, 16, 16, 1, 1, 3,
             Amat, Bmat, Cbuf, nullptr, nullptr);
    run_bn(stream, actC, actB, stats, stats2, B, 128, 4 * 256, actB, 1);
  }

  // ---- primary caps: conv 128->512 + per-batch LayerNorm ----
  run_conv(stream, w_primary, b_primary, actB, actA, B, 128, 4, 512, 16, 16, 1, 1, 3,
           Amat, Bmat, Cbuf, nullptr, nullptr);
  long per = 512L * 4 * 256;
  k_ln_stats<<<B, 256, 0, stream>>>(actA, stats, stats2, per, 1e-5f);
  k_ln_apply<<<cdiv(B * per, 256), 256, 0, stream>>>(actA, actB, stats, stats2, per, B * per);
  // actB is now caps (8,32,16,4,16,16) == (B'=256, d_in=16, 4, 16, 16)

  // ---- 5 capsule layers ----
  float* cur = actB;
  float* nxt = actA;
  int Hc = 16, Wc = 16;
  const int specs[5][3] = {{32, 2, 0}, {32, 1, 0}, {32, 1, 0}, {32, 1, 1}, {10, 1, 0}};
  for (int l = 0; l < 5; ++l) {
    int n_out = specs[l][0], st = specs[l][1], pd = specs[l][2];
    int Hn, Wn;
    // conv on (B*n_in=256, d_in=16, 4, Hc, Wc); result stays in Cbuf
    run_conv(stream, caps_w[l], nullptr, cur, nullptr, B * 32, 16, 4, n_out * 16,
             Hc, Wc, st, pd, 3, Amat, Bmat, Cbuf, &Hn, &Wn);
    int HWn = Hn * Wn;
    long nblocks = (long)B * n_out * 4 * HWn;
    k_routing<<<(unsigned)nblocks, 32, 0, stream>>>(Cbuf, caps_b[l], nxt, n_out, B, HWn);
    long sTot = (long)B * n_out * 16 * HWn;
    k_squash<<<cdiv(sTot, 256), 256, 0, stream>>>(nxt, sTot, HWn);
    float* tmp = cur; cur = nxt; nxt = tmp;
    Hc = Hn; Wc = Wn;
  }

  // ---- final per-class max-over-rotation norms: (8,10) ----
  k_final<<<1, 128, 0, stream>>>(cur, out, B);
}
